// GAEMI_10024453669136
// MI455X (gfx1250) — matrix operand order repaired and checked by reading, not compiled, not run on hardware
//
#include <hip/hip_runtime.h>
#include <hip/hip_bf16.h>

typedef __attribute__((ext_vector_type(16))) __bf16 v16bf;
typedef __attribute__((ext_vector_type(8)))  __bf16 v8bf;
typedef __attribute__((ext_vector_type(8)))  float  v8f;

#define NNODES 8192
#define NEDGES 262144
#define INDIM  512
#define H1DIM  256
#define H2DIM  128

// ---------------------------------------------------------------- utilities

__global__ void gae_zero_f32(float* __restrict__ p, int n) {
  int i = blockIdx.x * blockDim.x + threadIdx.x;
  if (i < n) p[i] = 0.0f;
}

__global__ void gae_deg(const int* __restrict__ src, const int* __restrict__ dst,
                        float* __restrict__ deg_out, float* __restrict__ deg_in, int E) {
  int e = blockIdx.x * blockDim.x + threadIdx.x;
  if (e < E) {
    atomicAdd(&deg_out[src[e]], 1.0f);
    atomicAdd(&deg_in[dst[e]], 1.0f);
  }
}

__global__ void gae_norms(const float* __restrict__ deg_out, const float* __restrict__ deg_in,
                          float* __restrict__ nsrc, float* __restrict__ ndst, int n) {
  int i = blockIdx.x * blockDim.x + threadIdx.x;
  if (i < n) {
    nsrc[i] = rsqrtf(fmaxf(deg_out[i], 1.0f));
    ndst[i] = rsqrtf(fmaxf(deg_in[i], 1.0f));
  }
}

// out[i] = bf16( x[i] * norm[row] ), row-major [rows x 2^logC]
__global__ void gae_scale_cvt(const float* __restrict__ x, const float* __restrict__ norm,
                              __bf16* __restrict__ out, int total, int logC) {
  int i = blockIdx.x * blockDim.x + threadIdx.x;
  if (i < total) out[i] = (__bf16)(x[i] * norm[i >> logC]);
}

// W is [K x N] row-major fp32 -> Wt is [N x K] row-major bf16
__global__ void gae_transpose_cvt(const float* __restrict__ W, __bf16* __restrict__ Wt,
                                  int K, int N) {
  int i = blockIdx.x * blockDim.x + threadIdx.x;
  if (i < K * N) {
    int k = i / N;
    int n = i - k * N;
    Wt[(size_t)n * K + k] = (__bf16)W[i];
  }
}

// agg[dst[e]][j] += H[src[e]][j]   (F = 1<<logF features per row)
__global__ void gae_scatter_add(const float* __restrict__ H, const int* __restrict__ src,
                                const int* __restrict__ dst, float* __restrict__ agg,
                                int E, int logF) {
  unsigned t = blockIdx.x * blockDim.x + threadIdx.x;
  unsigned total = ((unsigned)E) << logF;
  if (t >= total) return;
  unsigned e = t >> logF;
  unsigned j = t & ((1u << logF) - 1u);
  float v = H[((size_t)src[e] << logF) + j];
  unsafeAtomicAdd(&agg[((size_t)dst[e] << logF) + j], v);
}

// x1s = bf16( relu(agg*ndst + b) * nsrc )  -- fused layer-1 epilogue + layer-2 input prep
__global__ void gae_finish1(const float* __restrict__ agg, const float* __restrict__ ndst,
                            const float* __restrict__ b, const float* __restrict__ nsrc,
                            __bf16* __restrict__ out, int total, int logF) {
  int i = blockIdx.x * blockDim.x + threadIdx.x;
  if (i < total) {
    int node = i >> logF;
    int j = i & ((1 << logF) - 1);
    float v = fmaxf(agg[i] * ndst[node] + b[j], 0.0f);
    out[i] = (__bf16)(v * nsrc[node]);
  }
}

// emb = agg*ndst + b (fp32 to d_out)  and  embbf = bf16(emb) for the logits GEMM
__global__ void gae_finish2(const float* __restrict__ agg, const float* __restrict__ ndst,
                            const float* __restrict__ b, float* __restrict__ emb,
                            __bf16* __restrict__ embbf, int total, int logF) {
  int i = blockIdx.x * blockDim.x + threadIdx.x;
  if (i < total) {
    int node = i >> logF;
    int j = i & ((1 << logF) - 1);
    float v = agg[i] * ndst[node] + b[j];
    emb[i] = v;
    embbf[i] = (__bf16)v;
  }
}

// ---------------------------------------------------------------- WMMA GEMM
// C[M x N] = A[M x K] * Bt[N x K]^T   (both operands bf16, K-contiguous)
// Wave tile: 16 x 64 (4 x v_wmma_f32_16x16x32_bf16 per k-step, A frag reused 4x)
// EPI: 0 = plain fp32 store, 1 = sigmoid

__device__ __forceinline__ v16bf gae_cat8(v8bf lo, v8bf hi) {
  return __builtin_shufflevector(lo, hi, 0, 1, 2, 3, 4, 5, 6, 7,
                                          8, 9, 10, 11, 12, 13, 14, 15);
}

template <int EPI>
__global__ __launch_bounds__(256) void gae_wmma_gemm_nt(
    const __bf16* __restrict__ A, const __bf16* __restrict__ Bt,
    float* __restrict__ C, int M, int N, int K) {
  const int lane  = threadIdx.x & 31;
  const int wave  = threadIdx.x >> 5;
  const int tilesN = N >> 6;                       // 64-wide wave tiles
  const int tile   = blockIdx.x * 8 + wave;        // 8 waves per block
  if (tile >= (M >> 4) * tilesN) return;           // uniform per-wave exit (EXEC stays all-1)
  const int tm   = tile / tilesN;
  const int tn   = tile - tm * tilesN;
  const int half = lane >> 4;                      // 0: lanes 0-15, 1: lanes 16-31
  const int l15  = lane & 15;

  v8f acc0 = {}, acc1 = {}, acc2 = {}, acc3 = {};

  // A frag (16x32 bf16): lane half=0 holds K {0..15}, half=1 holds K {16..31}
  const __bf16* pA = A + (size_t)(tm * 16 + l15) * K + half * 16;
  // B frag (32x16 bf16): lane = column (l15), 16 contiguous K values per half
  const __bf16* pB = Bt + (size_t)(tn * 64 + l15) * K + half * 16;
  const size_t K16 = (size_t)K * 16;

  for (int k0 = 0; k0 < K; k0 += 32) {
    v8bf alo = *(const v8bf*)(pA + k0);
    v8bf ahi = *(const v8bf*)(pA + k0 + 8);
    v16bf a  = gae_cat8(alo, ahi);
    v16bf b0 = *(const v16bf*)(pB + k0);
    v16bf b1 = *(const v16bf*)(pB + K16 + k0);
    v16bf b2 = *(const v16bf*)(pB + 2 * K16 + k0);
    v16bf b3 = *(const v16bf*)(pB + 3 * K16 + k0);
    acc0 = __builtin_amdgcn_wmma_f32_16x16x32_bf16(false, a, false, b0, (short)0, acc0, false, false);
    acc1 = __builtin_amdgcn_wmma_f32_16x16x32_bf16(false, a, false, b1, (short)0, acc1, false, false);
    acc2 = __builtin_amdgcn_wmma_f32_16x16x32_bf16(false, a, false, b2, (short)0, acc2, false, false);
    acc3 = __builtin_amdgcn_wmma_f32_16x16x32_bf16(false, a, false, b3, (short)0, acc3, false, false);
  }

  // C layout: element i -> row = i + 8*half, col = l15
  float* pC = C + (size_t)(tm * 16 + half * 8) * N + (size_t)tn * 64 + l15;
#pragma unroll
  for (int t = 0; t < 4; ++t) {
    v8f acc = (t == 0) ? acc0 : (t == 1) ? acc1 : (t == 2) ? acc2 : acc3;
#pragma unroll
    for (int i = 0; i < 8; ++i) {
      float v = acc[i];
      if (EPI == 1) v = 1.0f / (1.0f + __expf(-v));
      pC[(size_t)i * N + t * 16] = v;
    }
  }
}

// ---------------------------------------------------------------- launcher

extern "C" void kernel_launch(void* const* d_in, const int* in_sizes, int n_in,
                              void* d_out, int out_size, void* d_ws, size_t ws_size,
                              hipStream_t stream) {
  const float* features = (const float*)d_in[0];
  const int*   src      = (const int*)d_in[1];
  const int*   dst      = (const int*)d_in[2];
  const float* W1       = (const float*)d_in[3];
  const float* b1       = (const float*)d_in[4];
  const float* W2       = (const float*)d_in[5];
  const float* b2       = (const float*)d_in[6];

  char* ws = (char*)d_ws;
  size_t off = 0;
  auto alloc = [&](size_t bytes) -> char* {
    char* p = ws + off;
    off = (off + bytes + 255) & ~(size_t)255;
    return p;
  };

  float*  deg_out = (float*)alloc((size_t)NNODES * 4);
  float*  deg_in  = (float*)alloc((size_t)NNODES * 4);
  float*  nsrc    = (float*)alloc((size_t)NNODES * 4);
  float*  ndst    = (float*)alloc((size_t)NNODES * 4);
  __bf16* xbf     = (__bf16*)alloc((size_t)NNODES * INDIM * 2);
  __bf16* w1t     = (__bf16*)alloc((size_t)H1DIM * INDIM * 2);
  float*  h1      = (float*)alloc((size_t)NNODES * H1DIM * 4);
  float*  agg1    = (float*)alloc((size_t)NNODES * H1DIM * 4);
  __bf16* x1s     = (__bf16*)alloc((size_t)NNODES * H1DIM * 2);
  __bf16* w2t     = (__bf16*)alloc((size_t)H2DIM * H1DIM * 2);
  float*  h2      = (float*)alloc((size_t)NNODES * H2DIM * 4);
  float*  agg2    = (float*)alloc((size_t)NNODES * H2DIM * 4);
  __bf16* embbf   = (__bf16*)alloc((size_t)NNODES * H2DIM * 2);

  float* emb_out = (float*)d_out;                       // [8192 x 128]
  float* logits  = emb_out + (size_t)NNODES * H2DIM;    // [8192 x 8192]

  const int B = 256;

  // degrees + norms
  gae_zero_f32<<<(NNODES + B - 1) / B, B, 0, stream>>>(deg_out, NNODES);
  gae_zero_f32<<<(NNODES + B - 1) / B, B, 0, stream>>>(deg_in, NNODES);
  gae_deg<<<(NEDGES + B - 1) / B, B, 0, stream>>>(src, dst, deg_out, deg_in, NEDGES);
  gae_norms<<<(NNODES + B - 1) / B, B, 0, stream>>>(deg_out, deg_in, nsrc, ndst, NNODES);

  // layer 1: (x*nsrc) @ W1, then scatter, then relu epilogue fused with layer-2 input prep
  gae_scale_cvt<<<((NNODES * INDIM) + B - 1) / B, B, 0, stream>>>(
      features, nsrc, xbf, NNODES * INDIM, 9);
  gae_transpose_cvt<<<((INDIM * H1DIM) + B - 1) / B, B, 0, stream>>>(W1, w1t, INDIM, H1DIM);
  {
    int tiles = (NNODES / 16) * (H1DIM / 64);  // 2048
    gae_wmma_gemm_nt<0><<<tiles / 8, B, 0, stream>>>(xbf, w1t, h1, NNODES, H1DIM, INDIM);
  }
  gae_zero_f32<<<((NNODES * H1DIM) + B - 1) / B, B, 0, stream>>>(agg1, NNODES * H1DIM);
  gae_scatter_add<<<(unsigned)(((size_t)NEDGES << 8) / B), B, 0, stream>>>(
      h1, src, dst, agg1, NEDGES, 8);
  gae_finish1<<<((NNODES * H1DIM) + B - 1) / B, B, 0, stream>>>(
      agg1, ndst, b1, nsrc, x1s, NNODES * H1DIM, 8);

  // layer 2
  gae_transpose_cvt<<<((H1DIM * H2DIM) + B - 1) / B, B, 0, stream>>>(W2, w2t, H1DIM, H2DIM);
  {
    int tiles = (NNODES / 16) * (H2DIM / 64);  // 1024
    gae_wmma_gemm_nt<0><<<tiles / 8, B, 0, stream>>>(x1s, w2t, h2, NNODES, H2DIM, H1DIM);
  }
  gae_zero_f32<<<((NNODES * H2DIM) + B - 1) / B, B, 0, stream>>>(agg2, NNODES * H2DIM);
  gae_scatter_add<<<(unsigned)(((size_t)NEDGES << 7) / B), B, 0, stream>>>(
      h2, src, dst, agg2, NEDGES, 7);
  gae_finish2<<<((NNODES * H2DIM) + B - 1) / B, B, 0, stream>>>(
      agg2, ndst, b2, emb_out, embbf, NNODES * H2DIM, 7);

  // logits = sigmoid(emb @ emb.T): both fragments K-contiguous from the same bf16 buffer
  {
    int tiles = (NNODES / 16) * (NNODES / 64);  // 65536
    gae_wmma_gemm_nt<1><<<tiles / 8, B, 0, stream>>>(embbf, embbf, logits, NNODES, NNODES, H2DIM);
  }
}
